// HybridQuantumClassifier_78374563217606
// MI455X (gfx1250) — matrix operand order repaired and checked by reading, not compiled, not run on hardware
//
#include <hip/hip_runtime.h>

#define NQ      14
#define DIM     (1 << NQ)        // 16384 amplitudes
#define M14     0x3FFFu
#define NLAYERS 4
#define TPB     256
#define NCLS    10
#define NBATCH  1024

typedef __attribute__((ext_vector_type(16))) _Float16 v16h;
typedef __attribute__((ext_vector_type(8)))  float    v8f;

// ---------------------------------------------------------------------------
// CNOT-ring as a linear map over GF(2)^14 on basis-index bits.
// Forward:  c_k = b_0^...^b_k (k>=1),  c_0 = b_1^...^b_13
// Inverse:  b_0 = c_0^c_13, b_1 = c_0^c_1^c_13, b_k = c_{k-1}^c_k (k>=2)
// ---------------------------------------------------------------------------
__device__ __forceinline__ unsigned ring_fwd(unsigned b) {
  unsigned t = b;
  t ^= t << 1; t ^= t << 2; t ^= t << 4; t ^= t << 8;   // prefix XOR
  t &= M14;
  return (t & 0x3FFEu) | (((t >> 13) ^ b) & 1u);
}
__device__ __forceinline__ unsigned ring_inv(unsigned c) {
  unsigned s = (c ^ (c << 1)) & M14;
  if ((c >> 13) & 1u) s ^= 3u;
  return s;
}

// Symmetric RX pair update on interleaved complex: u' = c*u - i*s*v ; v' = c*v - i*s*u
__device__ __forceinline__ void rx_pair(float2& u, float2& v, float c, float s) {
  const float ax = c * u.x + s * v.y;
  const float ay = c * u.y - s * v.x;
  const float bx = c * v.x + s * u.y;
  const float by = c * v.y - s * u.x;
  u.x = ax; u.y = ay; v.x = bx; v.y = by;
}

// ---------------------------------------------------------------------------
// One full-state sweep applying G fused RX gates with XOR-pair masks h[0..G-1].
// Group of 2^G amplitudes = coset of span{h}; element m = j ^ X[m] where X[m]
// XORs the masks selected by bits of m, so gate g pairs (m, m^(1<<g)) with
// compile-time register indices. Pivot bits (Gaussian elimination, uniform
// scalar work) give a unique zero-bit representative j per coset.
// Amplitudes are float2 (interleaved complex) -> single ds_load_b64 each.
// ---------------------------------------------------------------------------
template<int G>
__device__ __forceinline__ void rx_sweep(float2* __restrict__ samp,
                                         const int t,
                                         const unsigned* h,
                                         const float* gc, const float* gs)
{
  constexpr int NE = 1 << G;

  // eliminated basis -> G distinct pivot bit positions (all wave-uniform)
  unsigned he[G];
  int piv[G];
#pragma unroll
  for (int g = 0; g < G; ++g) {
    unsigned v = h[g];
    for (int k = 0; k < g; ++k)
      if ((v >> piv[k]) & 1u) v ^= he[k];
    he[g] = v;
    piv[g] = __builtin_ctz(v);
  }
  int ps[G];
#pragma unroll
  for (int g = 0; g < G; ++g) ps[g] = piv[g];
#pragma unroll
  for (int a = 0; a < G - 1; ++a)
#pragma unroll
    for (int q = 0; q < G - 1 - a; ++q)
      if (ps[q] > ps[q + 1]) { const int tv = ps[q]; ps[q] = ps[q + 1]; ps[q + 1] = tv; }

  // XOR combos of the ORIGINAL masks (uniform)
  unsigned X[NE];
  X[0] = 0u;
#pragma unroll
  for (int m = 1; m < NE; ++m) X[m] = X[m & (m - 1)] ^ h[__builtin_ctz((unsigned)m)];

  for (int p = t; p < (DIM >> G); p += TPB) {
    unsigned j = (unsigned)p;                         // insert zeros at pivots
#pragma unroll
    for (int g = 0; g < G; ++g)
      j = ((j >> ps[g]) << (ps[g] + 1)) | (j & ((1u << ps[g]) - 1));

    float2 amp[NE];
#pragma unroll
    for (int m = 0; m < NE; ++m) amp[m] = samp[j ^ X[m]];

#pragma unroll
    for (int g = 0; g < G; ++g)
#pragma unroll
      for (int m = 0; m < NE; ++m)
        if (!((m >> g) & 1))
          rx_pair(amp[m], amp[m | (1 << g)], gc[g], gs[g]);

#pragma unroll
    for (int m = 0; m < NE; ++m) samp[j ^ X[m]] = amp[m];
  }
}

// ---------------------------------------------------------------------------
// Kernel 1: one workgroup per sample; 14-qubit statevector in 128 KB LDS.
// CNOT rings are virtual (index relabeling); RX gates are symmetric XOR-mask
// pair updates fused 5+5+4 per layer; layer-0 RX folded into init.
// ---------------------------------------------------------------------------
__global__ __launch_bounds__(TPB)
void qsim_kernel(const float* __restrict__ x,      // [B, NQ]
                 const float* __restrict__ qp,     // [NLAYERS, NQ]
                 float* __restrict__ qout)         // [B, NQ]
{
  extern __shared__ float smem[];
  float2* samp = (float2*)smem;       // [DIM] interleaved complex, 128 KB

  const int b = blockIdx.x;
  const int t = threadIdx.x;

  // --- init: RY(x) product state with layer-0 RX folded in (still product) ---
  float u0r[NQ], u0i[NQ], u1r[NQ], u1i[NQ];
#pragma unroll
  for (int w = 0; w < NQ; ++w) {
    const float ax = 0.5f * x[b * NQ + w];
    const float cx = __cosf(ax), sx = __sinf(ax);
    const float th = 0.5f * qp[w];                 // layer 0 weights
    const float gc = __cosf(th), gs = __sinf(th);
    u0r[w] = gc * cx;  u0i[w] = -gs * sx;          // RX * (cx, sx): top
    u1r[w] = gc * sx;  u1i[w] = -gs * cx;          // bottom
  }
  for (int i = t; i < DIM; i += TPB) {
    float pr = 1.0f, pi = 0.0f;
#pragma unroll
    for (int w = 0; w < NQ; ++w) {
      const bool bit = (i >> w) & 1;
      const float ur = bit ? u1r[w] : u0r[w];
      const float ui = bit ? u1i[w] : u0i[w];
      const float nr = pr * ur - pi * ui;
      const float ni = pr * ui + pi * ur;
      pr = nr; pi = ni;
    }
    samp[i] = make_float2(pr, pi);
  }
  __syncthreads();

  // --- layers 1..3: three fused sweeps per layer (wires 0-4, 5-9, 10-13) ---
  for (int l = 1; l < NLAYERS; ++l) {
    unsigned h[5];
    float gc[5], gs[5];

    for (int base = 0; base < 10; base += 5) {
      for (int g = 0; g < 5; ++g) {
        const int w = base + g;
        unsigned mm = 1u << w;                      // h = F^{-l}(e_w)
        for (int r = 0; r < l; ++r) mm = ring_inv(mm);
        h[g] = mm;
        const float th = 0.5f * qp[l * NQ + w];
        gc[g] = __cosf(th); gs[g] = __sinf(th);
      }
      rx_sweep<5>(samp, t, h, gc, gs);
      __syncthreads();
    }
    for (int g = 0; g < 4; ++g) {
      const int w = 10 + g;
      unsigned mm = 1u << w;
      for (int r = 0; r < l; ++r) mm = ring_inv(mm);
      h[g] = mm;
      const float th = 0.5f * qp[l * NQ + w];
      gc[g] = __cosf(th); gs[g] = __sinf(th);
    }
    rx_sweep<4>(samp, t, h, gc, gs);
    __syncthreads();
  }

  // --- <Z_w> with virtual relabeling: sign = parity(j & m_w), m_w = rows of F^4
  unsigned m[NQ];
#pragma unroll
  for (int w = 0; w < NQ; ++w) m[w] = 0u;
#pragma unroll
  for (int jb = 0; jb < NQ; ++jb) {
    unsigned col = 1u << jb;
    col = ring_fwd(ring_fwd(ring_fwd(ring_fwd(col))));  // F^4 e_jb
#pragma unroll
    for (int w = 0; w < NQ; ++w) m[w] |= ((col >> w) & 1u) << jb;
  }

  float acc[NQ];
#pragma unroll
  for (int w = 0; w < NQ; ++w) acc[w] = 0.0f;
  for (int i = t; i < DIM; i += TPB) {
    const float2 a = samp[i];
    const float p2 = a.x * a.x + a.y * a.y;
#pragma unroll
    for (int w = 0; w < NQ; ++w)
      acc[w] += (__popc((unsigned)i & m[w]) & 1) ? -p2 : p2;
  }
  __syncthreads();
#pragma unroll
  for (int w = 0; w < NQ; ++w) smem[w * TPB + t] = acc[w];   // reuse LDS
  __syncthreads();
  for (int off = TPB / 2; off > 0; off >>= 1) {
    if (t < off) {
#pragma unroll
      for (int w = 0; w < NQ; ++w)
        smem[w * TPB + t] += smem[w * TPB + t + off];
    }
    __syncthreads();
  }
  if (t < NQ) qout[b * NQ + t] = smem[t * TPB];
}

// ---------------------------------------------------------------------------
// Kernel 2: logits = q_out @ fc_w^T + fc_b via V_WMMA_F32_16X16X32_F16.
// One wave per 16-row batch tile; K=14 zero-padded to 32, N=10 padded to 16.
// ---------------------------------------------------------------------------
__global__ __launch_bounds__(32)
void fc_wmma_kernel(const float* __restrict__ qout,   // [B, NQ]
                    const float* __restrict__ fcw,    // [NCLS, NQ]
                    const float* __restrict__ fcb,    // [NCLS]
                    float* __restrict__ out)          // [B, NCLS]
{
  const int tile = blockIdx.x;        // 16 batch rows per tile
  const int lane = threadIdx.x;       // wave32
  const int g    = lane >> 4;         // lane-half selects K group
  const int nm   = lane & 15;         // A: row M ; B & C/D: col N

  v16h a, bm;
#pragma unroll
  for (int e = 0; e < 16; ++e) {
    // A (16x32 f16): lanes 0-15: K=0..7 (v0-3), 16..23 (v4-7); lanes 16-31: +8
    const int ka = ((e < 8) ? e : e + 8) + g * 8;
    a[e] = (ka < NQ) ? (_Float16)qout[(tile * 16 + nm) * NQ + ka]
                     : (_Float16)0.0f;
    // B (32x16 f16): col N = lane&15; lanes 0-15: K=0..7, 16..23; lanes 16-31: +8
    const int kb = ((e < 8) ? e : e + 8) + g * 8;
    bm[e] = (kb < NQ && nm < NCLS) ? (_Float16)fcw[nm * NQ + kb]
                                   : (_Float16)0.0f;
  }

  v8f c = {};
  c = __builtin_amdgcn_wmma_f32_16x16x32_f16(
      /*neg_a=*/false, a, /*neg_b=*/false, bm,
      /*c_mod=*/(short)0, c, /*reuse_a=*/false, /*reuse_b=*/false);

  // C/D f32 16x16: VGPR r -> M = r + 8*(lane>=16), N = lane&15
  const float bias = (nm < NCLS) ? fcb[nm] : 0.0f;
#pragma unroll
  for (int r = 0; r < 8; ++r) {
    const int M = r + g * 8;
    if (nm < NCLS)
      out[(tile * 16 + M) * NCLS + nm] = c[r] + bias;
  }
}

// ---------------------------------------------------------------------------
extern "C" void kernel_launch(void* const* d_in, const int* in_sizes, int n_in,
                              void* d_out, int out_size, void* d_ws, size_t ws_size,
                              hipStream_t stream) {
  const float* x   = (const float*)d_in[0];   // [1024,14]
  const float* qp  = (const float*)d_in[1];   // [4,14]
  const float* fcw = (const float*)d_in[2];   // [10,14]
  const float* fcb = (const float*)d_in[3];   // [10]
  float* out  = (float*)d_out;                // [1024,10]
  float* qout = (float*)d_ws;                 // [1024,14] scratch

  const size_t shmem = (size_t)(2 * DIM) * sizeof(float);  // 128 KB
  qsim_kernel<<<NBATCH, TPB, shmem, stream>>>(x, qp, qout);
  fc_wmma_kernel<<<NBATCH / 16, 32, 0, stream>>>(qout, fcw, fcb, out);
}
